// GPT_69904887710628
// MI455X (gfx1250) — matrix operand order repaired and checked by reading, not compiled, not run on hardware
//
#include <hip/hip_runtime.h>
#include <hip/hip_bf16.h>

// ---------------------------------------------------------------------------
// Model dims (fixed by the reference)
// ---------------------------------------------------------------------------
#define LAYERS 7
#define DMODEL 1024
#define NHEAD  8
#define DHEAD  128
#define FFDIM  4096
#define VOCAB  27000
#define EDIM   960
#define PDIM   64
#define BATCH  4
#define SEQ    512
#define BT     (BATCH * SEQ)   // 2048 token rows

// ---------------------------------------------------------------------------
// WMMA types (CDNA5 gfx1250, wave32)
// ---------------------------------------------------------------------------
typedef __attribute__((ext_vector_type(16))) __bf16        v16bf;
typedef __attribute__((ext_vector_type(8)))  float         v8f;
typedef __attribute__((ext_vector_type(4)))  unsigned int  v4u;

union FragBF { v4u u[2]; v16bf v; };   // 256-bit fragment = 8 VGPRs

__device__ __forceinline__ unsigned short f2bf(float f) {
  // round-to-nearest-even fp32 -> bf16
  unsigned int u = __float_as_uint(f);
  u += 0x7FFFu + ((u >> 16) & 1u);
  return (unsigned short)(u >> 16);
}

// ---------------------------------------------------------------------------
// Generic batched WMMA GEMM:  C = scale * (A @ B) + bias   (+relu / causal)
//   A: [M,K] f32 row-major (lda), read once, converted to bf16 in LDS
//   B: f32; if !(flags&1): [K,N] row-major (ldb = row stride)
//           if  (flags&1): [N,K] row-major (ldb = row stride)  ("B^T in mem")
//   C: [M,N] f32 (ldc)
//   Batch z = blockIdx.z; outer = z/innerCount, inner = z%innerCount,
//   pointers advanced by the given element strides.
//   flags: 1 = B transposed in memory, 2 = ReLU, 4 = causal mask (n>m -> -1e9)
// ---------------------------------------------------------------------------
#define TILE_M 128
#define TILE_N 128
#define TILE_K 64
#define PITCH  72   // ushort pitch -> 144B rows (16B aligned for b128 LDS loads)

__global__ __launch_bounds__(256) void wmma_gemm_kernel(
    const float* __restrict__ A, const float* __restrict__ B,
    const float* __restrict__ bias, float* __restrict__ C,
    int M, int N, int K, int lda, int ldb, int ldc,
    long aOuter, long aInner, long bOuter, long bInner, long cOuter, long cInner,
    int innerCount, float scale, int flags)
{
  __shared__ __align__(16) unsigned short As[TILE_M][PITCH];
  __shared__ __align__(16) unsigned short Bs[TILE_N][PITCH];   // [n][k] (transposed)

  const int tid  = threadIdx.x;
  const int lane = tid & 31;
  const int wave = tid >> 5;     // 8 waves
  const int wm   = wave >> 2;    // 0..1  (64-row band)
  const int wn   = wave & 3;     // 0..3  (32-col band)

  const int m0 = blockIdx.y * TILE_M;
  const int n0 = blockIdx.x * TILE_N;

  const int z  = blockIdx.z;
  const int ob = z / innerCount;
  const int ib = z % innerCount;
  A += (long)ob * aOuter + (long)ib * aInner;
  B += (long)ob * bOuter + (long)ib * bInner;
  C += (long)ob * cOuter + (long)ib * cInner;

  v8f acc[4][2] = {};

  const bool btrans = (flags & 1) != 0;

  for (int k0 = 0; k0 < K; k0 += TILE_K) {
    // ---- stage A tile (128 x 64), fp32 -> bf16; 32 elems / thread ----
    {
      const int m  = tid >> 1;
      const int kc = (tid & 1) * 32;
      const float* ap = A + (size_t)(m0 + m) * lda + k0 + kc;
      #pragma unroll
      for (int i = 0; i < 32; i += 4) {
        float4 f = *(const float4*)(ap + i);
        As[m][kc + i + 0] = f2bf(f.x);
        As[m][kc + i + 1] = f2bf(f.y);
        As[m][kc + i + 2] = f2bf(f.z);
        As[m][kc + i + 3] = f2bf(f.w);
      }
    }
    // ---- stage B tile into transposed LDS Bs[n][k] ----
    if (btrans) {
      const int n  = tid >> 1;
      const int kc = (tid & 1) * 32;
      if (n0 + n < N) {
        const float* bp = B + (size_t)(n0 + n) * ldb + k0 + kc;
        #pragma unroll
        for (int i = 0; i < 32; i += 4) {
          float4 f = *(const float4*)(bp + i);
          Bs[n][kc + i + 0] = f2bf(f.x);
          Bs[n][kc + i + 1] = f2bf(f.y);
          Bs[n][kc + i + 2] = f2bf(f.z);
          Bs[n][kc + i + 3] = f2bf(f.w);
        }
      } else {
        #pragma unroll
        for (int i = 0; i < 32; ++i) Bs[n][kc + i] = 0;
      }
    } else {
      const int kr = tid >> 2;          // 0..63
      const int nc = (tid & 3) * 32;
      const float* bp = B + (size_t)(k0 + kr) * ldb + n0 + nc;
      #pragma unroll
      for (int i = 0; i < 32; ++i) {
        const int n = nc + i;
        Bs[n][kr] = (n0 + n < N) ? f2bf(bp[i]) : (unsigned short)0;
      }
    }
    __syncthreads();

    // ---- 2 WMMA k-steps per LDS tile, 8 WMMAs each ----
    #pragma unroll
    for (int kk = 0; kk < TILE_K; kk += 32) {
      FragBF a[4], b[2];
      // A frag layout (16x32 bf16): lanes 0-15 -> K 0..7 / 16..23,
      //                             lanes16-31 -> K 8..15 / 24..31
      const int khA = (lane >> 4) * 8;
      #pragma unroll
      for (int mt = 0; mt < 4; ++mt) {
        const int r = wm * 64 + mt * 16 + (lane & 15);
        a[mt].u[0] = *(const v4u*)&As[r][kk + khA];
        a[mt].u[1] = *(const v4u*)&As[r][kk + 16 + khA];
      }
      // B frag layout (32x16 bf16): lanes 0-15 -> K 0..7 / 16..23, lanes16-31 -> K 8..15 / 24..31
      const int khB = (lane >> 4) * 8;
      #pragma unroll
      for (int nt = 0; nt < 2; ++nt) {
        const int n = wn * 32 + nt * 16 + (lane & 15);
        b[nt].u[0] = *(const v4u*)&Bs[n][kk + khB];
        b[nt].u[1] = *(const v4u*)&Bs[n][kk + 16 + khB];
      }
      #pragma unroll
      for (int mt = 0; mt < 4; ++mt)
        #pragma unroll
        for (int nt = 0; nt < 2; ++nt)
          acc[mt][nt] = __builtin_amdgcn_wmma_f32_16x16x32_bf16(
              false, a[mt].v, false, b[nt].v, (short)0, acc[mt][nt],
              false, false);
    }
    __syncthreads();
  }

  // ---- epilogue: scale + bias + mask/relu, one f32 store per element ----
  const bool relu   = (flags & 2) != 0;
  const bool causal = (flags & 4) != 0;
  #pragma unroll
  for (int mt = 0; mt < 4; ++mt) {
    #pragma unroll
    for (int nt = 0; nt < 2; ++nt) {
      const int n = n0 + wn * 32 + nt * 16 + (lane & 15);
      if (n >= N) continue;
      const float bv = bias ? bias[n] : 0.0f;
      #pragma unroll
      for (int r = 0; r < 8; ++r) {
        const int m = m0 + wm * 64 + mt * 16 + r + (lane >> 4) * 8;
        float v = acc[mt][nt][r] * scale + bv;
        if (causal && n > m) v = -1e9f;
        if (relu)            v = fmaxf(v, 0.0f);
        C[(size_t)m * ldc + n] = v;
      }
    }
  }
}

// ---------------------------------------------------------------------------
// Embedding gather + positional concat: h[b*T+t] = [meaning_W.T[tok], pos[t]]
// ---------------------------------------------------------------------------
__global__ __launch_bounds__(256) void embed_kernel(
    const int* __restrict__ x, const float* __restrict__ mW,
    const float* __restrict__ mB, const float* __restrict__ pos,
    float* __restrict__ h)
{
  const int row = blockIdx.x;           // b*SEQ + t
  const int t   = row & (SEQ - 1);
  const int tok = x[row];
  for (int c = threadIdx.x; c < DMODEL; c += 256) {
    float v;
    if (c < EDIM) v = mW[(size_t)c * VOCAB + tok] + mB[c];   // meaning_W is [E,V]
    else          v = pos[t * PDIM + (c - EDIM)];
    h[(size_t)row * DMODEL + c] = v;
  }
}

// ---------------------------------------------------------------------------
// In-place row softmax over 512 elements (one block per row, wave32 reductions)
// ---------------------------------------------------------------------------
__global__ __launch_bounds__(256) void softmax512_kernel(float* __restrict__ S)
{
  __shared__ float red[8];
  float* s = S + (size_t)blockIdx.x * SEQ;
  const int tid = threadIdx.x;
  float a = s[tid], b = s[tid + 256];

  float m = fmaxf(a, b);
  #pragma unroll
  for (int o = 16; o > 0; o >>= 1) m = fmaxf(m, __shfl_xor(m, o, 32));
  if ((tid & 31) == 0) red[tid >> 5] = m;
  __syncthreads();
  float mm = red[0];
  #pragma unroll
  for (int i = 1; i < 8; ++i) mm = fmaxf(mm, red[i]);
  __syncthreads();

  a = __expf(a - mm);
  b = __expf(b - mm);
  float sum = a + b;
  #pragma unroll
  for (int o = 16; o > 0; o >>= 1) sum += __shfl_xor(sum, o, 32);
  if ((tid & 31) == 0) red[tid >> 5] = sum;
  __syncthreads();
  float tot = 0.0f;
  #pragma unroll
  for (int i = 0; i < 8; ++i) tot += red[i];
  const float inv = 1.0f / tot;
  s[tid]       = a * inv;
  s[tid + 256] = b * inv;
}

// ---------------------------------------------------------------------------
// Fused residual + LayerNorm over D=1024: Y = LN(X + Dlt) * g + b  (Y may == X)
// ---------------------------------------------------------------------------
__global__ __launch_bounds__(256) void layernorm_kernel(
    const float* __restrict__ X, const float* __restrict__ Dlt,
    const float* __restrict__ g, const float* __restrict__ b,
    float* __restrict__ Y)
{
  __shared__ float red[8];
  const int row = blockIdx.x, tid = threadIdx.x;
  const float* x = X   + (size_t)row * DMODEL;
  const float* d = Dlt + (size_t)row * DMODEL;

  float v[4], s = 0.0f;
  #pragma unroll
  for (int i = 0; i < 4; ++i) { v[i] = x[tid + 256 * i] + d[tid + 256 * i]; s += v[i]; }

  #pragma unroll
  for (int o = 16; o > 0; o >>= 1) s += __shfl_xor(s, o, 32);
  if ((tid & 31) == 0) red[tid >> 5] = s;
  __syncthreads();
  float tot = 0.0f;
  #pragma unroll
  for (int i = 0; i < 8; ++i) tot += red[i];
  const float mu = tot * (1.0f / DMODEL);
  __syncthreads();

  float vs = 0.0f;
  #pragma unroll
  for (int i = 0; i < 4; ++i) { const float c = v[i] - mu; vs += c * c; }
  #pragma unroll
  for (int o = 16; o > 0; o >>= 1) vs += __shfl_xor(vs, o, 32);
  if ((tid & 31) == 0) red[tid >> 5] = vs;
  __syncthreads();
  float vtot = 0.0f;
  #pragma unroll
  for (int i = 0; i < 8; ++i) vtot += red[i];
  const float inv = rsqrtf(vtot * (1.0f / DMODEL) + 1e-5f);

  #pragma unroll
  for (int i = 0; i < 4; ++i) {
    const int c = tid + 256 * i;
    Y[(size_t)row * DMODEL + c] = (v[i] - mu) * inv * g[c] + b[c];
  }
}

// ---------------------------------------------------------------------------
// Host orchestration
// ---------------------------------------------------------------------------
enum { F_BTRANS = 1, F_RELU = 2, F_CAUSAL = 4 };

static void gemm(hipStream_t s, const float* A, const float* B, const float* bias,
                 float* C, int M, int N, int K, int lda, int ldb, int ldc,
                 long aO, long aI, long bO, long bI, long cO, long cI,
                 int innerCount, int batches, float scale, int flags)
{
  dim3 grid((N + TILE_N - 1) / TILE_N, (M + TILE_M - 1) / TILE_M, batches);
  hipLaunchKernelGGL(wmma_gemm_kernel, grid, dim3(256), 0, s,
                     A, B, bias, C, M, N, K, lda, ldb, ldc,
                     aO, aI, bO, bI, cO, cI,
                     innerCount > 0 ? innerCount : 1, scale, flags);
}

extern "C" void kernel_launch(void* const* d_in, const int* in_sizes, int n_in,
                              void* d_out, int out_size, void* d_ws, size_t ws_size,
                              hipStream_t stream)
{
  (void)in_sizes; (void)n_in; (void)out_size; (void)ws_size;

  const int*   x         = (const int*)  d_in[0];
  const float* meaning_W = (const float*)d_in[1];
  const float* meaning_b = (const float*)d_in[2];
  const float* pos_emb   = (const float*)d_in[3];
  const float* Wqkv      = (const float*)d_in[4];
  const float* bqkv      = (const float*)d_in[5];
  const float* Wo        = (const float*)d_in[6];
  const float* bo        = (const float*)d_in[7];
  const float* ln1_g     = (const float*)d_in[8];
  const float* ln1_b     = (const float*)d_in[9];
  const float* Wff1      = (const float*)d_in[10];
  const float* bff1      = (const float*)d_in[11];
  const float* Wff2      = (const float*)d_in[12];
  const float* bff2      = (const float*)d_in[13];
  const float* ln2_g     = (const float*)d_in[14];
  const float* ln2_b     = (const float*)d_in[15];
  const float* out_W     = (const float*)d_in[16];
  const float* out_b     = (const float*)d_in[17];
  float* logits = (float*)d_out;

  // workspace bump allocator (total ~118 MB)
  char* ws = (char*)d_ws;
  size_t off = 0;
  auto alloc = [&](size_t bytes) -> void* {
    void* p = ws + off;
    off = (off + bytes + 255) & ~(size_t)255;
    return p;
  };
  float* h      = (float*)alloc((size_t)BT * DMODEL * 4);             // [2048,1024]
  float* qkv    = (float*)alloc((size_t)BT * 3 * DMODEL * 4);         // [2048,3072]
  float* scores = (float*)alloc((size_t)BATCH * NHEAD * SEQ * SEQ * 4); // [32,512,512]
  float* ctx    = (float*)alloc((size_t)BT * DMODEL * 4);             // [2048,1024]
  float* delta  = (float*)alloc((size_t)BT * DMODEL * 4);             // [2048,1024]
  float* ffb    = (float*)alloc((size_t)BT * FFDIM * 4);              // [2048,4096]

  const float attn_scale = 0.088388347648318447f;  // 1/sqrt(128)

  // 1) embedding gather + positional concat
  hipLaunchKernelGGL(embed_kernel, dim3(BT), dim3(256), 0, stream,
                     x, meaning_W, meaning_b, pos_emb, h);

  // 2) transformer layers
  for (int l = 0; l < LAYERS; ++l) {
    const float* wqkv_l = Wqkv + (size_t)l * DMODEL * 3 * DMODEL;
    const float* wo_l   = Wo   + (size_t)l * DMODEL * DMODEL;
    const float* wf1_l  = Wff1 + (size_t)l * DMODEL * FFDIM;
    const float* wf2_l  = Wff2 + (size_t)l * FFDIM * DMODEL;

    // qkv = h @ Wqkv + bqkv                       [2048,3072]
    gemm(stream, h, wqkv_l, bqkv + (size_t)l * 3 * DMODEL, qkv,
         BT, 3 * DMODEL, DMODEL, DMODEL, 3 * DMODEL, 3 * DMODEL,
         0, 0, 0, 0, 0, 0, 1, 1, 1.0f, 0);

    // scores[b,h] = scale * Q_bh @ K_bh^T  (+ causal for layer 0)  batched 32x
    gemm(stream,
         /*A=Q*/ qkv,            /*B=K (as [N=tok][K=dh])*/ qkv + DMODEL,
         /*bias*/ nullptr,       /*C*/ scores,
         SEQ, SEQ, DHEAD, 3 * DMODEL, 3 * DMODEL, SEQ,
         (long)SEQ * 3 * DMODEL, DHEAD,                 // A: per-b, per-h
         (long)SEQ * 3 * DMODEL, DHEAD,                 // B: per-b, per-h
         (long)NHEAD * SEQ * SEQ, (long)SEQ * SEQ,      // C: per-b, per-h
         NHEAD, BATCH * NHEAD, attn_scale,
         F_BTRANS | (l == 0 ? F_CAUSAL : 0));

    // softmax rows (in place)
    hipLaunchKernelGGL(softmax512_kernel, dim3(BATCH * NHEAD * SEQ), dim3(256),
                       0, stream, scores);

    // ctx[b,:,h*128..] = P_bh @ V_bh               batched 32x
    gemm(stream,
         /*A=P*/ scores,         /*B=V*/ qkv + 2 * DMODEL,
         nullptr, ctx,
         SEQ, DHEAD, SEQ, SEQ, 3 * DMODEL, DMODEL,
         (long)NHEAD * SEQ * SEQ, (long)SEQ * SEQ,      // A: per-b, per-h
         (long)SEQ * 3 * DMODEL, DHEAD,                 // B: per-b, per-h
         (long)SEQ * DMODEL, DHEAD,                     // C: per-b, per-h
         NHEAD, BATCH * NHEAD, 1.0f, 0);

    // delta = ctx @ Wo + bo
    gemm(stream, ctx, wo_l, bo + (size_t)l * DMODEL, delta,
         BT, DMODEL, DMODEL, DMODEL, DMODEL, DMODEL,
         0, 0, 0, 0, 0, 0, 1, 1, 1.0f, 0);

    // h = LN(h + delta)
    hipLaunchKernelGGL(layernorm_kernel, dim3(BT), dim3(256), 0, stream,
                       h, delta, ln1_g + (size_t)l * DMODEL,
                       ln1_b + (size_t)l * DMODEL, h);

    // ffb = relu(h @ Wff1 + bff1)                 [2048,4096]
    gemm(stream, h, wf1_l, bff1 + (size_t)l * FFDIM, ffb,
         BT, FFDIM, DMODEL, DMODEL, FFDIM, FFDIM,
         0, 0, 0, 0, 0, 0, 1, 1, 1.0f, F_RELU);

    // delta = ffb @ Wff2 + bff2
    gemm(stream, ffb, wf2_l, bff2 + (size_t)l * DMODEL, delta,
         BT, DMODEL, FFDIM, FFDIM, DMODEL, DMODEL,
         0, 0, 0, 0, 0, 0, 1, 1, 1.0f, 0);

    // h = LN(h + delta)
    hipLaunchKernelGGL(layernorm_kernel, dim3(BT), dim3(256), 0, stream,
                       h, delta, ln2_g + (size_t)l * DMODEL,
                       ln2_b + (size_t)l * DMODEL, h);
  }

  // 3) logits = h @ out_W^T + out_b   (out_W is [V,D] row-major -> B_TRANS)
  gemm(stream, h, out_W, out_b, logits,
       BT, VOCAB, DMODEL, DMODEL, DMODEL, VOCAB,
       0, 0, 0, 0, 0, 0, 1, 1, 1.0f, F_BTRANS);
}
